// ExpertChoiceFF_58506044506432
// MI455X (gfx1250) — compile-verified, matrix operand order checked by reading, not run on hardware
//
#include <hip/hip_runtime.h>
#include <hip/hip_bf16.h>

// out = relu(x @ w1 + b1) @ w2 + b2
// x:[8192,1024] f32, w1:[1024,4096], w2:[4096,1024], out:[8192,1024] f32
// bf16 WMMA (v_wmma_f32_16x16x32_bf16) with f32 accumulation.
// Operand staging via Tensor Data Mover (tensor_load_to_lds) with hardware
// LDS row padding, double-buffered, TENSORcnt-synchronized.

typedef __bf16 bf16;
typedef __bf16 v8bf  __attribute__((ext_vector_type(8)));
typedef __bf16 v16bf __attribute__((ext_vector_type(16)));
typedef float  v4f   __attribute__((ext_vector_type(4)));
typedef float  v8f   __attribute__((ext_vector_type(8)));
typedef unsigned int v4u __attribute__((ext_vector_type(4)));
typedef int    v4i   __attribute__((ext_vector_type(4)));
typedef int    v8i   __attribute__((ext_vector_type(8)));

#define M_TOT 8192   // B*S
#define D_DIM 1024
#define W_DIM 4096

// ---------------- prep: f32 -> bf16 elementwise ----------------
__global__ __launch_bounds__(256)
void cvt_f32_bf16(const float* __restrict__ in, bf16* __restrict__ out) {
  int i = (blockIdx.x * 256 + threadIdx.x) * 4;
  v4f v = *(const v4f*)(in + i);
  out[i + 0] = (bf16)v.x;
  out[i + 1] = (bf16)v.y;
  out[i + 2] = (bf16)v.z;
  out[i + 3] = (bf16)v.w;
}

// ---------------- prep: w[K,N] f32 -> wt[N,K] bf16 (LDS-tiled) ----------------
__global__ __launch_bounds__(1024)
void transpose_cvt(const float* __restrict__ w, bf16* __restrict__ wt, int K, int N) {
  __shared__ float tile[32][33];
  int n0 = blockIdx.x * 32, k0 = blockIdx.y * 32;
  int tx = threadIdx.x, ty = threadIdx.y;
  tile[ty][tx] = w[(size_t)(k0 + ty) * N + (n0 + tx)];
  __syncthreads();
  wt[(size_t)(n0 + ty) * K + (k0 + tx)] = (bf16)tile[tx][ty];
}

// ---------------- TDM helpers ----------------
__device__ __forceinline__ unsigned lds_off(const void* p) {
  // Flat LDS addresses carry the LDS byte offset in the low 32 bits
  // (aperture check uses ADDR[63:32] only).
  return (unsigned)(unsigned long long)(uintptr_t)p;
}

// 2-D tile load: global (row-major, row stride = strideElems bf16) -> LDS,
// with HW padding of 4 dwords (16B) after every 16 dwords (64B) so the LDS
// row stride becomes 40 halfs — matching the WMMA fragment loader below.
__device__ __forceinline__ void tdm_load_2d(unsigned ldsByteOff, const bf16* g,
                                            unsigned tileRows, unsigned strideElems) {
  unsigned long long ga = (unsigned long long)(uintptr_t)g;
  v4u g0;
  g0.x = 1u;                                    // count=1 valid descriptor
  g0.y = ldsByteOff;                            // lds_addr (bytes)
  g0.z = (unsigned)(ga & 0xffffffffu);          // global_addr[31:0]
  g0.w = (unsigned)((ga >> 32) & 0x01ffffffu)   // global_addr[56:32]
         | (2u << 30);                          // type = 2 ("image")
  v8i g1;
  // data_size=1 (2B) <<16 | pad_enable <<20 | pad_interval=3 (16 dw) <<22
  // | pad_amount=3 (4 dw) <<25
  g1[0] = (int)((1u << 16) | (1u << 20) | (3u << 22) | (3u << 25));
  unsigned td0 = strideElems;                   // tensor_dim0 (elements)
  unsigned td1 = 1u << 20;                      // tensor_dim1: large, no OOB
  g1[1] = (int)((td0 & 0xffffu) << 16);         // tensor_dim0[15:0]
  g1[2] = (int)((td0 >> 16) | ((td1 & 0xffffu) << 16));
  g1[3] = (int)((td1 >> 16) | (32u << 16));     // tile_dim0 = 32 (K chunk)
  g1[4] = (int)(tileRows & 0xffffu);            // tile_dim1 = rows; tile_dim2=0
  g1[5] = (int)strideElems;                     // tensor_dim0_stride[31:0]
  g1[6] = 0;                                    // stride[47:32], dim1_stride lo
  g1[7] = 0;
  v4i z = {0, 0, 0, 0};
#if __clang_major__ >= 23
  v8i z8 = {0, 0, 0, 0, 0, 0, 0, 0};
  __builtin_amdgcn_tensor_load_to_lds(g0, g1, z, z, z8, 0);
#else
  __builtin_amdgcn_tensor_load_to_lds(g0, g1, z, z, 0);
#endif
}

// ---------------- main GEMM: C[M,N] = A[M,K] @ Bt[N,K]^T + bias ----------------
// Block tile 256x128, BK=32, 8 wave32 in 4x2; wave tile 64x64 = 4x4 WMMA accums.
// TDM double-buffered operand staging: wave 0 issues tensor loads for tile i+1,
// all waves compute tile i from LDS, wave 0 waits TENSORcnt, one barrier.
template <bool RELU_BF16>
__global__ __launch_bounds__(256)
void gemm_wmma_bf16(const bf16* __restrict__ A, const bf16* __restrict__ Bt,
                    const float* __restrict__ bias, void* __restrict__ Cout,
                    int N, int K) {
  constexpr int LDT = 40;  // padded LDS row stride (halfs) = TDM pad result
  __shared__ __align__(16) bf16 As[2][256 * LDT];   // 2 x 20 KB
  __shared__ __align__(16) bf16 Bs[2][128 * LDT];   // 2 x 10 KB

  const int tid  = threadIdx.x;
  const int lane = tid & 31;
  const int wid  = tid >> 5;       // 0..7
  const int wm   = wid >> 1;       // 0..3 -> M offset wm*64
  const int wn   = wid & 1;        // 0..1 -> N offset wn*64
  const int lr   = lane & 15;      // row within 16x16 fragment
  const int hi   = lane >> 4;      // K-half select (ISA 7.12.2 16-bit layout)
  const int bm   = blockIdx.y, bn = blockIdx.x;

  const bf16* aG = A  + (size_t)(bm * 256) * K;
  const bf16* bG = Bt + (size_t)(bn * 128) * K;

  const v8f vzero = {0.f, 0.f, 0.f, 0.f, 0.f, 0.f, 0.f, 0.f};
  v8f acc[4][4];
#pragma unroll
  for (int im = 0; im < 4; ++im)
#pragma unroll
    for (int in = 0; in < 4; ++in) acc[im][in] = vzero;

  // Preload buffer 0 (wave 0 issues DMA; TENSORcnt is per-wave).
  if (wid == 0) {
    tdm_load_2d(lds_off(&As[0][0]), aG, 256, (unsigned)K);
    tdm_load_2d(lds_off(&Bs[0][0]), bG, 128, (unsigned)K);
    __builtin_amdgcn_s_wait_tensorcnt(0);
  }
  __syncthreads();

  int cur = 0;
  for (int kt = 0; kt < K; kt += 32) {
    // Kick off next tile's DMA into the other buffer (safe: last barrier
    // guarantees everyone finished reading it).
    if (kt + 32 < K && wid == 0) {
      tdm_load_2d(lds_off(&As[cur ^ 1][0]), aG + (kt + 32), 256, (unsigned)K);
      tdm_load_2d(lds_off(&Bs[cur ^ 1][0]), bG + (kt + 32), 128, (unsigned)K);
    }

    const bf16* as = &As[cur][0];
    const bf16* bs = &Bs[cur][0];
    v16bf afrag[4], bfrag[4];
#pragma unroll
    for (int im = 0; im < 4; ++im) {
      int row = wm * 64 + im * 16 + lr;
      v8bf c0 = *(const v8bf*)&as[row * LDT + hi * 8];        // K = hi*8..+7
      v8bf c1 = *(const v8bf*)&as[row * LDT + hi * 8 + 16];   // K = 16+hi*8..+7
      afrag[im] = __builtin_shufflevector(c0, c1, 0, 1, 2, 3, 4, 5, 6, 7,
                                          8, 9, 10, 11, 12, 13, 14, 15);
    }
#pragma unroll
    for (int in = 0; in < 4; ++in) {
      int row = wn * 64 + in * 16 + lr;
      v8bf c0 = *(const v8bf*)&bs[row * LDT + hi * 8];
      v8bf c1 = *(const v8bf*)&bs[row * LDT + hi * 8 + 16];
      bfrag[in] = __builtin_shufflevector(c0, c1, 0, 1, 2, 3, 4, 5, 6, 7,
                                          8, 9, 10, 11, 12, 13, 14, 15);
    }
#pragma unroll
    for (int im = 0; im < 4; ++im)
#pragma unroll
      for (int in = 0; in < 4; ++in)
        acc[im][in] = __builtin_amdgcn_wmma_f32_16x16x32_bf16(
            /*neg_a=*/false, afrag[im], /*neg_b=*/false, bfrag[in],
            /*c_mod=*/(short)0, acc[im][in], /*reuse_a=*/false, /*reuse_b=*/false);

    // Next buffer must be resident before anyone reads it next iteration.
    if (wid == 0) __builtin_amdgcn_s_wait_tensorcnt(0);
    __syncthreads();
    cur ^= 1;
  }

  // Epilogue. C/D layout: lanes 0-15 -> M=j, lanes 16-31 -> M=8+j; N = lane%16.
#pragma unroll
  for (int im = 0; im < 4; ++im)
#pragma unroll
    for (int in = 0; in < 4; ++in) {
      int n = bn * 128 + wn * 64 + in * 16 + lr;
      float bv = bias[n];
#pragma unroll
      for (int j = 0; j < 8; ++j) {
        int m = bm * 256 + wm * 64 + im * 16 + hi * 8 + j;
        float v = acc[im][in][j] + bv;
        if (RELU_BF16) {
          v = fmaxf(v, 0.0f);
          ((bf16*)Cout)[(size_t)m * N + n] = (bf16)v;
        } else {
          ((float*)Cout)[(size_t)m * N + n] = v;
        }
      }
    }
}

extern "C" void kernel_launch(void* const* d_in, const int* in_sizes, int n_in,
                              void* d_out, int out_size, void* d_ws, size_t ws_size,
                              hipStream_t stream) {
  const float* x  = (const float*)d_in[0];  // [8192,1024]
  const float* w1 = (const float*)d_in[1];  // [1024,4096]
  const float* b1 = (const float*)d_in[2];  // [4096]
  const float* w2 = (const float*)d_in[3];  // [4096,1024]
  const float* b2 = (const float*)d_in[4];  // [1024]

  char* ws  = (char*)d_ws;
  bf16* xb  = (bf16*)(ws);                  // 16 MB  [8192,1024]
  bf16* w1t = (bf16*)(ws + (16u << 20));    //  8 MB  [4096,1024]  (w1^T)
  bf16* w2t = (bf16*)(ws + (24u << 20));    //  8 MB  [1024,4096]  (w2^T)
  bf16* h   = (bf16*)(ws + (32u << 20));    // 64 MB  [8192,4096]

  cvt_f32_bf16<<<(M_TOT * D_DIM) / 1024, 256, 0, stream>>>(x, xb);
  transpose_cvt<<<dim3(W_DIM / 32, D_DIM / 32), dim3(32, 32), 0, stream>>>(w1, w1t, D_DIM, W_DIM);
  transpose_cvt<<<dim3(D_DIM / 32, W_DIM / 32), dim3(32, 32), 0, stream>>>(w2, w2t, W_DIM, D_DIM);

  // h = relu(x @ w1 + b1), stored bf16
  gemm_wmma_bf16<true><<<dim3(W_DIM / 128, M_TOT / 256), 256, 0, stream>>>(
      xb, w1t, b1, (void*)h, W_DIM, D_DIM);
  // out = h @ w2 + b2, stored f32
  gemm_wmma_bf16<false><<<dim3(D_DIM / 128, M_TOT / 256), 256, 0, stream>>>(
      h, w2t, b2, d_out, D_DIM, W_DIM);
}
